// Recommender_2_43868795961766
// MI455X (gfx1250) — matrix operand order repaired and checked by reading, not compiled, not run on hardware
//
#include <hip/hip_runtime.h>
#include <cstddef>
#include <cstdint>

// ---------------------------------------------------------------------------
// CDNA5 / gfx1250 WMMA recommender implementation (wave32, bf16 WMMA, f32 acc)
// ---------------------------------------------------------------------------

typedef __bf16 bf16_t;
typedef __attribute__((ext_vector_type(16))) __bf16 v16bf;
typedef __attribute__((ext_vector_type(8)))  float  v8f;

union FragAB { v16bf v; unsigned int u[8]; uint4 q[2]; };
union FragC  { v8f   v; float        f[8]; };
union U32H   { unsigned int u; __bf16 h[2]; };

__device__ __forceinline__ float sigmoidf_(float x) {
    return 1.0f / (1.0f + __expf(-x));
}

// ---------------------------------------------------------------------------
// f32 -> bf16 conversion (weights)
// ---------------------------------------------------------------------------
__global__ void cvt_f32_bf16_kernel(const float* __restrict__ s,
                                    bf16_t* __restrict__ d, int n) {
    for (int i = blockIdx.x * blockDim.x + threadIdx.x; i < n;
         i += gridDim.x * blockDim.x)
        d[i] = (bf16_t)s[i];
}

// ---------------------------------------------------------------------------
// Embedding gathers
// ---------------------------------------------------------------------------
__global__ void gather_seq_kernel(const int* __restrict__ idx,
                                  const float* __restrict__ emb,
                                  bf16_t* __restrict__ out, int nbt) {
    int total = nbt * 128;
    for (int i = blockIdx.x * blockDim.x + threadIdx.x; i < total;
         i += gridDim.x * blockDim.x) {
        int bt = i >> 7, d = i & 127;
        out[i] = (bf16_t)emb[(size_t)idx[bt] * 128 + d];
    }
}

__global__ void gather_uin_kernel(const int* __restrict__ uid,
                                  const float* __restrict__ uemb,
                                  const float* __restrict__ udata,
                                  bf16_t* __restrict__ out) {
    int total = 1024 * 96;
    for (int i = blockIdx.x * blockDim.x + threadIdx.x; i < total;
         i += gridDim.x * blockDim.x) {
        int b = i / 96, j = i % 96;
        float v = (j < 64) ? uemb[(size_t)uid[b] * 64 + j]
                           : udata[(size_t)b * 32 + (j - 64)];
        out[i] = (bf16_t)v;
    }
}

// ---------------------------------------------------------------------------
// Generic tiled WMMA GEMM:  C[MxN] = A[MxK] * B  (+bias0+bias1, opt ReLU)
//   BT=true : B is NxK row-major ("NT", contiguous per-lane B fragments)
//   BT=false: B is KxN row-major ("NN", transposed through LDS)
// Block tile 64x32 (8 waves, each one 16x16 WMMA tile), K in steps of 32.
// ---------------------------------------------------------------------------
template <bool BT, bool RELU, bool OBF16>
__global__ void __launch_bounds__(256)
gemm_bf16_kernel(const bf16_t* __restrict__ A, int lda,
                 const bf16_t* __restrict__ Bm, int ldb,
                 const float* __restrict__ bias0, const float* __restrict__ bias1,
                 float* __restrict__ Cf, bf16_t* __restrict__ Cb, int ldc,
                 int K) {
    __shared__ bf16_t As[64][36];
    __shared__ bf16_t Bs[32][36];

    const int tid  = threadIdx.x;
    const int wave = tid >> 5, lane = tid & 31;
    const int mwg  = blockIdx.y * 64, nwg = blockIdx.x * 32;
    const int msub = (wave >> 1) * 16, nsub = (wave & 1) * 16;
    const int lrow = lane & 15, lhi = lane >> 4;

    FragC c;
#pragma unroll
    for (int j = 0; j < 8; ++j) c.f[j] = 0.0f;

    const int KT = K >> 5;
#pragma unroll 1
    for (int kt = 0; kt < KT; ++kt) {
        const int k0 = kt * 32;
        // stage A tile (64 x 32) as u32 pairs
#pragma unroll
        for (int i = 0; i < 4; ++i) {
            int id = tid + i * 256;
            int r = id >> 4, cp = id & 15;
            unsigned int w =
                *(const unsigned int*)(A + (size_t)(mwg + r) * lda + k0 + cp * 2);
            *(unsigned int*)(&As[r][cp * 2]) = w;
        }
        if (!BT) {
            // stage B tile (32k x 32n), transposed to Bs[n][k]
#pragma unroll
            for (int i = 0; i < 2; ++i) {
                int id = tid + i * 256;
                int k = id >> 4, np = id & 15;
                U32H w;
                w.u = *(const unsigned int*)(Bm + (size_t)(k0 + k) * ldb + nwg +
                                             np * 2);
                Bs[np * 2][k]     = w.h[0];
                Bs[np * 2 + 1][k] = w.h[1];
            }
        }
        __syncthreads();

        FragAB a, b;
#pragma unroll
        for (int j = 0; j < 8; ++j) {  // contiguous k layout (k = 16*lhi + s)
            int kk = lhi * 16 + j * 2;
            a.u[j] = *(const unsigned int*)(&As[msub + lrow][kk]);
        }
        if (BT) {
            const uint4* bp = (const uint4*)(Bm +
                (size_t)(nwg + nsub + lrow) * ldb + k0 + lhi * 16);
            b.q[0] = bp[0];
            b.q[1] = bp[1];
        } else {
#pragma unroll
            for (int j = 0; j < 8; ++j)
                b.u[j] =
                    *(const unsigned int*)(&Bs[nsub + lrow][lhi * 16 + j * 2]);
        }
        c.v = __builtin_amdgcn_wmma_f32_16x16x32_bf16(false, a.v, false, b.v,
                                                      (short)0, c.v, false,
                                                      false);
        __syncthreads();
    }

    const int n = nwg + nsub + lrow;
    float bv = 0.0f;
    if (bias0) bv += bias0[n];
    if (bias1) bv += bias1[n];
#pragma unroll
    for (int j = 0; j < 8; ++j) {
        int m   = mwg + msub + lhi * 8 + j;
        float v = c.f[j] + bv;
        if (RELU) v = fmaxf(v, 0.0f);
        if (OBF16) Cb[(size_t)m * ldc + n] = (bf16_t)v;
        else       Cf[(size_t)m * ldc + n] = v;
    }
}

// ---------------------------------------------------------------------------
// LSTM layer: each WG owns 16 batch rows, h/c live in LDS across all T steps.
// Gates g = xg[:,t,:] + h @ Whh^T via WMMA (Whh is 1024x256 bf16, NT layout,
// L2-resident). Elementwise i/f/g/o update in LDS, optional seq/h_last out.
// Compute loops are kept rolled (#pragma unroll 1) so only one A/B fragment
// pair is live at a time -> no VGPR spills.
// ---------------------------------------------------------------------------
__global__ void __launch_bounds__(256)
lstm_layer_kernel(const float* __restrict__ xg, const bf16_t* __restrict__ Whh,
                  bf16_t* __restrict__ seq_out, bf16_t* __restrict__ hlast,
                  int T) {
    __shared__ bf16_t h_s[16][264];
    __shared__ float  c_s[16][256];
    __shared__ bf16_t g_s[16][1024];

    const int tid  = threadIdx.x;
    const int wave = tid >> 5, lane = tid & 31;
    const int lrow = lane & 15, lhi = lane >> 4;
    const int bb   = blockIdx.x * 16;

    for (int i = tid; i < 4096; i += 256) {
        int m = i >> 8, nn = i & 255;
        h_s[m][nn] = (bf16_t)0.0f;
        c_s[m][nn] = 0.0f;
    }
    __syncthreads();

#pragma unroll 1
    for (int t = 0; t < T; ++t) {
        // gates: 16 x 1024, each wave owns 128 gate columns (8 n-tiles)
#pragma unroll 1
        for (int nt = 0; nt < 8; ++nt) {
            const int nb = wave * 128 + nt * 16;
            FragC acc;
#pragma unroll
            for (int j = 0; j < 8; ++j) acc.f[j] = 0.0f;
#pragma unroll 1
            for (int kt = 0; kt < 8; ++kt) {
                FragAB a, b;
#pragma unroll
                for (int j = 0; j < 8; ++j) {
                    int kk = kt * 32 + lhi * 16 + j * 2;
                    a.u[j] = *(const unsigned int*)(&h_s[lrow][kk]);
                }
                const uint4* bp = (const uint4*)(Whh +
                    (size_t)(nb + lrow) * 256 + kt * 32 + lhi * 16);
                b.q[0] = bp[0];
                b.q[1] = bp[1];
                acc.v = __builtin_amdgcn_wmma_f32_16x16x32_bf16(
                    false, a.v, false, b.v, (short)0, acc.v, false, false);
            }
            const int n = nb + lrow;
            const float* xgp =
                xg + ((size_t)bb * T + t) * 1024 + n + (size_t)lhi * 8 * T * 1024;
#pragma unroll
            for (int j = 0; j < 8; ++j) {
                g_s[lhi * 8 + j][n] =
                    (bf16_t)(acc.f[j] + xgp[(size_t)j * T * 1024]);
            }
        }
        __syncthreads();
        // elementwise cell update
        for (int i = tid; i < 4096; i += 256) {
            int m = i >> 8, nn = i & 255;
            float gi = (float)g_s[m][nn];
            float gf = (float)g_s[m][nn + 256];
            float gg = (float)g_s[m][nn + 512];
            float go = (float)g_s[m][nn + 768];
            float cc = sigmoidf_(gf) * c_s[m][nn] + sigmoidf_(gi) * tanhf(gg);
            float hh = sigmoidf_(go) * tanhf(cc);
            c_s[m][nn] = cc;
            h_s[m][nn] = (bf16_t)hh;
            if (seq_out)
                seq_out[((size_t)(bb + m) * T + t) * 256 + nn] = (bf16_t)hh;
        }
        __syncthreads();
    }
    if (hlast) {
        for (int i = tid; i < 4096; i += 256) {
            int m = i >> 8, nn = i & 255;
            hlast[(size_t)(bb + m) * 256 + nn] = h_s[m][nn];
        }
    }
}

// ---------------------------------------------------------------------------
// Fused MoE head: per (expert e, 64-row batch tile):
//   he = relu(z @ W1[e] + b1[e]);  out[b,e] = he . W2[e] + b2[e]
// z tile staged once in LDS; W1 (bf16, 75 MB total, L2-resident) transposed
// through LDS per 32x32 block; dot with W2 fused in epilogue via ds_add_f32.
// Compute loops kept rolled to avoid VGPR spills.
// ---------------------------------------------------------------------------
__global__ void __launch_bounds__(256)
expert_moe_kernel(const bf16_t* __restrict__ z, const bf16_t* __restrict__ W1,
                  const float* __restrict__ b1, const float* __restrict__ W2,
                  const float* __restrict__ b2, float* __restrict__ out) {
    __shared__ bf16_t zs[64][392];
    __shared__ bf16_t Ws[32][36];
    __shared__ float  accum[64];

    const int tid  = threadIdx.x;
    const int wave = tid >> 5, lane = tid & 31;
    const int lrow = lane & 15, lhi = lane >> 4;
    const int msub = (wave >> 1) * 16, nsub = (wave & 1) * 16;
    const int bb   = blockIdx.x * 64;
    const int e    = blockIdx.y;

    for (int i = tid; i < 12288; i += 256) {  // 64 x 384 bf16 as u32 pairs
        int r = i / 192, cp = i % 192;
        *(unsigned int*)(&zs[r][cp * 2]) =
            *(const unsigned int*)(z + (size_t)(bb + r) * 384 + cp * 2);
    }
    if (tid < 64) accum[tid] = 0.0f;
    __syncthreads();

    float priv[8];
#pragma unroll
    for (int j = 0; j < 8; ++j) priv[j] = 0.0f;

#pragma unroll 1
    for (int nt = 0; nt < 48; ++nt) {  // EH = 1536 = 48 * 32
        FragC c;
#pragma unroll
        for (int j = 0; j < 8; ++j) c.f[j] = 0.0f;
#pragma unroll 1
        for (int kt = 0; kt < 12; ++kt) {  // EIN = 384 = 12 * 32
#pragma unroll
            for (int i = 0; i < 2; ++i) {
                int id = tid + i * 256;
                int k = id >> 4, np = id & 15;
                U32H w;
                w.u = *(const unsigned int*)(W1 +
                                             ((size_t)e * 384 + kt * 32 + k) *
                                                 1536 +
                                             nt * 32 + np * 2);
                Ws[np * 2][k]     = w.h[0];
                Ws[np * 2 + 1][k] = w.h[1];
            }
            __syncthreads();
            FragAB a, b;
#pragma unroll
            for (int j = 0; j < 8; ++j) {
                int kk = kt * 32 + lhi * 16 + j * 2;
                a.u[j] = *(const unsigned int*)(&zs[msub + lrow][kk]);
            }
#pragma unroll
            for (int j = 0; j < 8; ++j)
                b.u[j] =
                    *(const unsigned int*)(&Ws[nsub + lrow][lhi * 16 + j * 2]);
            c.v = __builtin_amdgcn_wmma_f32_16x16x32_bf16(
                false, a.v, false, b.v, (short)0, c.v, false, false);
            __syncthreads();
        }
        const int   n   = nt * 32 + nsub + lrow;
        const float bia = b1[(size_t)e * 1536 + n];
        const float w2  = W2[(size_t)e * 1536 + n];
#pragma unroll
        for (int j = 0; j < 8; ++j) {
            float hv = fmaxf(c.f[j] + bia, 0.0f);
            priv[j] += hv * w2;
        }
    }
#pragma unroll
    for (int j = 0; j < 8; ++j) {
        int m = msub + lhi * 8 + j;
        atomicAdd(&accum[m], priv[j]);  // ds_add_f32
    }
    __syncthreads();
    if (tid < 64) out[(size_t)(bb + tid) * 64 + e] = accum[tid] + b2[e];
}

// ---------------------------------------------------------------------------
// Host launcher
// ---------------------------------------------------------------------------
static inline size_t align256(size_t x) { return (x + 255) & ~(size_t)255; }

extern "C" void kernel_launch(void* const* d_in, const int* in_sizes, int n_in,
                              void* d_out, int out_size, void* d_ws,
                              size_t ws_size, hipStream_t stream) {
    (void)in_sizes; (void)n_in; (void)out_size; (void)ws_size;

    const int*   ratings   = (const int*)d_in[0];
    const int*   user_ids  = (const int*)d_in[1];
    const float* user_data = (const float*)d_in[2];
    const float* rat_emb   = (const float*)d_in[3];
    const float* usr_emb   = (const float*)d_in[4];
    const float* W_ih0 = (const float*)d_in[5];
    const float* W_hh0 = (const float*)d_in[6];
    const float* b_ih0 = (const float*)d_in[7];
    const float* b_hh0 = (const float*)d_in[8];
    const float* W_ih1 = (const float*)d_in[9];
    const float* W_hh1 = (const float*)d_in[10];
    const float* b_ih1 = (const float*)d_in[11];
    const float* b_hh1 = (const float*)d_in[12];
    const float* r_W1 = (const float*)d_in[13];
    const float* r_b1 = (const float*)d_in[14];
    const float* r_W2 = (const float*)d_in[15];
    const float* r_b2 = (const float*)d_in[16];
    const float* u_W1 = (const float*)d_in[17];
    const float* u_b1 = (const float*)d_in[18];
    const float* u_W2 = (const float*)d_in[19];
    const float* u_b2 = (const float*)d_in[20];
    const float* exp_W1 = (const float*)d_in[21];
    const float* exp_b1 = (const float*)d_in[22];
    const float* exp_W2 = (const float*)d_in[23];
    const float* exp_b2 = (const float*)d_in[24];
    float* out = (float*)d_out;

    const int B = 1024, T = 50, BT = B * T;

    char* p = (char*)d_ws;
    auto alloc = [&](size_t bytes) { char* r = p; p += align256(bytes); return r; };

    bf16_t* wb_ih0 = (bf16_t*)alloc((size_t)1024 * 128 * 2);
    bf16_t* wb_hh0 = (bf16_t*)alloc((size_t)1024 * 256 * 2);
    bf16_t* wb_ih1 = (bf16_t*)alloc((size_t)1024 * 256 * 2);
    bf16_t* wb_hh1 = (bf16_t*)alloc((size_t)1024 * 256 * 2);
    bf16_t* wb_r1  = (bf16_t*)alloc((size_t)256 * 512 * 2);
    bf16_t* wb_r2  = (bf16_t*)alloc((size_t)512 * 256 * 2);
    bf16_t* wb_u1  = (bf16_t*)alloc((size_t)96 * 192 * 2);
    bf16_t* wb_u2  = (bf16_t*)alloc((size_t)192 * 128 * 2);
    bf16_t* wb_e1  = (bf16_t*)alloc((size_t)64 * 384 * 1536 * 2);
    bf16_t* seq0   = (bf16_t*)alloc((size_t)BT * 128 * 2);
    bf16_t* seq1   = (bf16_t*)alloc((size_t)BT * 256 * 2);
    float*  xg     = (float*)alloc((size_t)BT * 1024 * 4);  // reused by both layers
    bf16_t* hlast  = (bf16_t*)alloc((size_t)B * 256 * 2);
    bf16_t* uin    = (bf16_t*)alloc((size_t)B * 96 * 2);
    bf16_t* u1a    = (bf16_t*)alloc((size_t)B * 192 * 2);
    bf16_t* r1a    = (bf16_t*)alloc((size_t)B * 512 * 2);
    bf16_t* zcat   = (bf16_t*)alloc((size_t)B * 384 * 2);

    auto cvt = [&](const float* s, bf16_t* d, size_t n) {
        int blocks = (int)((n + 255) / 256);
        if (blocks > 4096) blocks = 4096;
        cvt_f32_bf16_kernel<<<blocks, 256, 0, stream>>>(s, d, (int)n);
    };
    cvt(W_ih0, wb_ih0, (size_t)1024 * 128);
    cvt(W_hh0, wb_hh0, (size_t)1024 * 256);
    cvt(W_ih1, wb_ih1, (size_t)1024 * 256);
    cvt(W_hh1, wb_hh1, (size_t)1024 * 256);
    cvt(r_W1,  wb_r1,  (size_t)256 * 512);
    cvt(r_W2,  wb_r2,  (size_t)512 * 256);
    cvt(u_W1,  wb_u1,  (size_t)96 * 192);
    cvt(u_W2,  wb_u2,  (size_t)192 * 128);
    cvt(exp_W1, wb_e1, (size_t)64 * 384 * 1536);

    gather_seq_kernel<<<4096, 256, 0, stream>>>(ratings, rat_emb, seq0, BT);
    gather_uin_kernel<<<384, 256, 0, stream>>>(user_ids, usr_emb, user_data, uin);

    // layer 0: xg = seq0 @ W_ih0^T + (b_ih0 + b_hh0); then recurrence -> seq1
    gemm_bf16_kernel<true, false, false><<<dim3(1024 / 32, BT / 64), 256, 0, stream>>>(
        seq0, 128, wb_ih0, 128, b_ih0, b_hh0, xg, nullptr, 1024, 128);
    lstm_layer_kernel<<<B / 16, 256, 0, stream>>>(xg, wb_hh0, seq1, nullptr, T);

    // layer 1: xg = seq1 @ W_ih1^T + (b_ih1 + b_hh1); recurrence -> hlast only
    gemm_bf16_kernel<true, false, false><<<dim3(1024 / 32, BT / 64), 256, 0, stream>>>(
        seq1, 256, wb_ih1, 256, b_ih1, b_hh1, xg, nullptr, 1024, 256);
    lstm_layer_kernel<<<B / 16, 256, 0, stream>>>(xg, wb_hh1, nullptr, hlast, T);

    // r branch: r1 = relu(h @ r_W1 + r_b1); r = r1 @ r_W2 + r_b2 -> z[:,128:384]
    gemm_bf16_kernel<false, true, true><<<dim3(512 / 32, B / 64), 256, 0, stream>>>(
        hlast, 256, wb_r1, 512, r_b1, nullptr, nullptr, r1a, 512, 256);
    gemm_bf16_kernel<false, false, true><<<dim3(256 / 32, B / 64), 256, 0, stream>>>(
        r1a, 512, wb_r2, 256, r_b2, nullptr, nullptr, zcat + 128, 384, 512);

    // u branch: u1 = relu(uin @ u_W1 + u_b1); u = u1 @ u_W2 + u_b2 -> z[:,0:128]
    gemm_bf16_kernel<false, true, true><<<dim3(192 / 32, B / 64), 256, 0, stream>>>(
        uin, 96, wb_u1, 192, u_b1, nullptr, nullptr, u1a, 192, 96);
    gemm_bf16_kernel<false, false, true><<<dim3(128 / 32, B / 64), 256, 0, stream>>>(
        u1a, 192, wb_u2, 128, u_b2, nullptr, nullptr, zcat, 384, 192);

    // fused MoE head -> out (1024 x 64 f32)
    expert_moe_kernel<<<dim3(B / 64, 64), 256, 0, stream>>>(
        zcat, wb_e1, exp_b1, exp_W2, exp_b2, out);
}
